// create_ssn_net_25941602468104
// MI455X (gfx1250) — matrix operand order repaired and checked by reading, not compiled, not run on hardware
//
#include <hip/hip_runtime.h>

// ---------------- problem constants (from reference) ----------------
#define BQ   2
#define HQ   256
#define WQ   256
#define KSP  256      // 16x16 superpixels
#define NITER 5
#define EPSQ 1e-10f

typedef __bf16 bf16t;
typedef __attribute__((ext_vector_type(16))) __bf16 v16bf;   // 32 bytes
typedef __attribute__((ext_vector_type(8)))  __bf16 v8bf;    // 16 bytes
typedef __attribute__((ext_vector_type(8)))  float  v8f;

__device__ __forceinline__ int clip16(int v) { return v < 0 ? 0 : (v > 15 ? 15 : v); }

// ================= bulk zero (halo init for padded activations) =================
__global__ void zero_k(uint4* __restrict__ p, int n16)
{
  int idx = blockIdx.x * blockDim.x + threadIdx.x;
  if (idx < n16) p[idx] = uint4{0u, 0u, 0u, 0u};
}

// ================= weight prep =================
// fp32 OIHW -> bf16 [CoutPad][9*Cpad], k = tap*Cpad + ci (tap-major, channel-minor:
// each 32-wide K block lies inside a single 3x3 tap).
__global__ void prep_weights_k(const float* __restrict__ w, bf16t* __restrict__ wb,
                               int Cout, int Cin, int Cpad, int CoutPad)
{
  int Kpad = 9 * Cpad;
  int idx = blockIdx.x * blockDim.x + threadIdx.x;
  if (idx >= CoutPad * Kpad) return;
  int m = idx / Kpad, k = idx % Kpad;
  int r = k / Cpad, ci = k % Cpad;
  float v = 0.f;
  if (m < Cout && ci < Cin) v = w[(m * Cin + ci) * 9 + r];
  wb[idx] = (bf16t)v;
}

// Fold conv bias + BN into per-channel scale/bias: y = conv*s + t
__global__ void prep_scalebias_k(const float* __restrict__ b, const float* __restrict__ g,
                                 const float* __restrict__ bt, const float* __restrict__ mean,
                                 const float* __restrict__ var,
                                 float* __restrict__ s, float* __restrict__ t,
                                 int Cout, int CoutPad, int has_bn)
{
  int m = blockIdx.x * blockDim.x + threadIdx.x;
  if (m >= CoutPad) return;
  if (m >= Cout) { s[m] = 1.f; t[m] = 0.f; return; }
  if (has_bn) {
    float sv = g[m] * rsqrtf(var[m] + 1e-5f);
    s[m] = sv;
    t[m] = (b[m] - mean[m]) * sv + bt[m];
  } else {
    s[m] = 1.f; t[m] = b[m];
  }
}

// ================= input / concat packing into padded bf16 NHWC =================
// x [B,5,H,W] f32 -> xp [B][H+2][W+2][32] bf16 interior (halo pre-zeroed)
__global__ void pack_x_k(const float* __restrict__ x, bf16t* __restrict__ xp)
{
  int idx = blockIdx.x * blockDim.x + threadIdx.x;
  if (idx >= BQ * HQ * WQ) return;
  int xq = idx % WQ; int t = idx / WQ; int y = t % HQ; int b = t / HQ;
  bf16t* o = xp + (((size_t)(b * (HQ + 2)) + y + 1) * (WQ + 2) + xq + 1) * 32;
  #pragma unroll
  for (int c = 0; c < 5; ++c)
    o[c] = (bf16t)x[(((size_t)b * 5 + c) * HQ + y) * WQ + xq];
}

// cat = [x(5)|c2(64)|up2(c4)(64)|up4(c6)(64)|pad] -> [B][258][258][224] interior
__global__ void build_cat_k(const float* __restrict__ x, const bf16t* __restrict__ c2,
                            const bf16t* __restrict__ c4, const bf16t* __restrict__ c6,
                            bf16t* __restrict__ cat)
{
  int idx = blockIdx.x * blockDim.x + threadIdx.x;
  if (idx >= BQ * HQ * WQ) return;
  int xq = idx % WQ; int t = idx / WQ; int y = t % HQ; int b = t / HQ;
  bf16t* o = cat + (((size_t)(b * 258) + y + 1) * 258 + xq + 1) * 224;
  #pragma unroll
  for (int c = 0; c < 5; ++c)
    o[c] = (bf16t)x[(((size_t)b * 5 + c) * HQ + y) * WQ + xq];
  const bf16t* s2 = c2 + (((size_t)(b * 258) + y + 1) * 258 + xq + 1) * 64;
  for (int c = 0; c < 64; ++c) o[5 + c] = s2[c];
  const bf16t* s4 = c4 + (((size_t)(b * 130) + (y >> 1) + 1) * 130 + (xq >> 1) + 1) * 64;
  for (int c = 0; c < 64; ++c) o[69 + c] = s4[c];
  const bf16t* s6 = c6 + (((size_t)(b * 66) + (y >> 2) + 1) * 66 + (xq >> 2) + 1) * 64;
  for (int c = 0; c < 64; ++c) o[133 + c] = s6[c];
  // channels 197..223 already zero from bulk zero-fill
}

// ================= implicit-GEMM 3x3 conv, padded bf16 NHWC, WMMA bf16 =================
// act: [B][H+2][W+2][CPAD] bf16 with zero halo -> no bounds checks; all tap/K offsets
// are compile-time immediates. One wave per block: 16 Cout x 16 consecutive pixels.
template <int CPAD, int H, int W, bool F32OUT>
__global__ __launch_bounds__(32)
void conv3x3_wmma_k(const bf16t* __restrict__ act, const bf16t* __restrict__ wb,
                    const float* __restrict__ scale, const float* __restrict__ bias,
                    void* __restrict__ outv,
                    int Cout, int relu, int outChOff, int outChTot)
{
  constexpr int KPAD = 9 * CPAD;
  constexpr int WP   = W + 2;
  const int lane    = threadIdx.x;
  const int b       = blockIdx.z;
  const int mt      = blockIdx.y * 16;
  const int pixBase = blockIdx.x * 16;      // 16 pixels within one row (W % 16 == 0)
  const int n       = lane & 15;
  const int hiHalf  = lane >> 4;
  const int abase   = hiHalf * 8;           // C M-offset (ISA layout)
  const int bbase   = hiHalf * 16;          // A/B K-offset (contiguous k = 16h + s)
  const int y0      = pixBase / W;
  const int x0      = pixBase % W;

  const bf16t* wrow = wb + (size_t)(mt + n) * KPAD + bbase;
  const bf16t* pb   = act + (((size_t)(b * (H + 2)) + y0 + 1) * WP + x0 + n + 1) * CPAD + bbase;

  v8f acc = {};
  #pragma unroll
  for (int r = 0; r < 9; ++r) {
    const int off = ((r / 3 - 1) * WP + (r % 3 - 1)) * CPAD;   // compile-time immediate
    #pragma unroll
    for (int cb = 0; cb < CPAD; cb += 32) {
      // A fragment: two 16B chunks of the weight row, contiguous K = 16h..16h+15 (immediate offsets)
      v8bf alo = *(const v8bf*)(wrow + r * CPAD + cb);
      v8bf ahi = *(const v8bf*)(wrow + r * CPAD + cb + 8);
      v16bf av = __builtin_shufflevector(alo, ahi,
                   0,1,2,3,4,5,6,7,8,9,10,11,12,13,14,15);
      // B fragment: 32 contiguous 32B-aligned bytes (immediate offset, halo = zeros)
      v16bf bv = *(const v16bf*)(pb + off + cb);
      acc = __builtin_amdgcn_wmma_f32_16x16x32_bf16(false, av, false, bv,
                                                    (short)0, acc, false, false);
    }
  }

  // ---- epilogue: fused BN scale/bias (+ReLU) ----
  if (F32OUT) {       // conv7 -> feat (unpadded NHWC f32)
    float* out = (float*)outv;
    const size_t pixIdx = (size_t)(b * H + y0) * W + x0 + n;
    #pragma unroll
    for (int rr = 0; rr < 8; ++rr) {
      int m = mt + rr + abase;               // C layout: lanes 16..31 hold M = rr+8
      if (m < Cout) {
        float v = acc[rr] * scale[m] + bias[m];
        if (relu) v = fmaxf(v, 0.f);
        out[pixIdx * outChTot + outChOff + m] = v;
      }
    }
  } else {            // padded bf16 NHWC, 8 contiguous channels as one 16B store
    bf16t* out = (bf16t*)outv;
    const size_t pixIdx = ((size_t)(b * (H + 2)) + y0 + 1) * WP + x0 + n + 1;
    v8bf ov;
    #pragma unroll
    for (int rr = 0; rr < 8; ++rr) {
      int m = mt + rr + abase;
      float v = acc[rr] * scale[m] + bias[m];
      if (relu) v = fmaxf(v, 0.f);
      ov[rr] = (bf16t)v;
    }
    *(v8bf*)(out + pixIdx * outChTot + mt + abase) = ov;
  }
}

// ================= maxpool 3x3 stride2 pad1, padded bf16 NHWC (64 ch) =================
// Input is post-ReLU (>= 0) so reading the zero halo never changes the max.
template <int Hin, int Win>
__global__ void maxpool_k(const bf16t* __restrict__ in, bf16t* __restrict__ out)
{
  constexpr int Ho = Hin / 2, Wo = Win / 2;
  int idx = blockIdx.x * blockDim.x + threadIdx.x;
  if (idx >= BQ * Ho * Wo * 64) return;
  int c = idx & 63; int t = idx >> 6;
  int wo = t % Wo; t /= Wo;
  int ho = t % Ho; int b = t / Ho;
  float m = -3.402823466e38f;
  #pragma unroll
  for (int dy = -1; dy <= 1; ++dy)
    #pragma unroll
    for (int dx = -1; dx <= 1; ++dx) {
      int y = 2 * ho + dy, x = 2 * wo + dx;           // padded coords stay in range
      m = fmaxf(m, (float)in[(((size_t)(b * (Hin + 2)) + y + 1) * (Win + 2) + x + 1) * 64 + c]);
    }
  out[(((size_t)(b * (Ho + 2)) + ho + 1) * (Wo + 2) + wo + 1) * 64 + c] = (bf16t)m;
}

// ================= copy x f32 into feat NHWC channels 0..4 =================
__global__ void copyx_k(const float* __restrict__ x, float* __restrict__ feat)
{
  int idx = blockIdx.x * blockDim.x + threadIdx.x;
  if (idx >= BQ * HQ * WQ * 5) return;
  int c = idx % 5; int p = idx / 5;            // p = b*H*W + pix
  int pix = p % (HQ * WQ); int b = p / (HQ * WQ);
  feat[(size_t)p * 20 + c] = x[((size_t)b * 5 + c) * HQ * WQ + pix];
}

// ================= initial superpixel mean (exact 16x16 blocks) =================
__global__ void spf_init_k(const float* __restrict__ feat, float* __restrict__ spf)
{
  __shared__ float red[256];
  int k = blockIdx.x, b = blockIdx.y, tid = threadIdx.x;
  int y = (k / 16) * 16 + tid / 16;
  int x = (k % 16) * 16 + tid % 16;
  size_t p = ((size_t)b * HQ + y) * WQ + x;
  for (int c = 0; c < 20; ++c) {
    red[tid] = feat[p * 20 + c];
    __syncthreads();
    for (int s = 128; s > 0; s >>= 1) { if (tid < s) red[tid] += red[tid + s]; __syncthreads(); }
    if (tid == 0) spf[(b * KSP + k) * 20 + c] = red[0] * (1.f / 256.f);
    __syncthreads();
  }
}

// ================= 9-neighbor softmax assignment =================
// feat NHWC [B,H,W,20]; assoc [B,H,W,9]
__global__ void assign_k(const float* __restrict__ feat, const float* __restrict__ spf,
                         float* __restrict__ assoc)
{
  int idx = blockIdx.x * blockDim.x + threadIdx.x;
  if (idx >= BQ * HQ * WQ) return;
  int x = idx % WQ; int t = idx / WQ; int y = t % HQ; int b = t / HQ;

  float f[20];
  const float* fp = feat + (size_t)idx * 20;
  #pragma unroll
  for (int c = 0; c < 20; ++c) f[c] = fp[c];

  int by = y >> 4, bx = x >> 4;
  float lg[9]; float mx = -3.402823466e38f;
  #pragma unroll
  for (int j = 0; j < 9; ++j) {
    int ny = by + j / 3 - 1, nx = bx + j % 3 - 1;
    bool bad = (ny < 0) | (ny >= 16) | (nx < 0) | (nx >= 16);
    int sp = clip16(ny) * 16 + clip16(nx);
    const float* sv = spf + (b * KSP + sp) * 20;
    float d = 0.f;
    #pragma unroll
    for (int c = 0; c < 20; ++c) { float df = f[c] - sv[c]; d += df * df; }
    float l = -d - (bad ? 1e16f : 0.f);
    lg[j] = l; mx = fmaxf(mx, l);
  }
  float e[9], sum = 0.f;
  #pragma unroll
  for (int j = 0; j < 9; ++j) { e[j] = __expf(lg[j] - mx); sum += e[j]; }
  float inv = 1.f / sum;
  float* ap = assoc + (size_t)idx * 9;
  #pragma unroll
  for (int j = 0; j < 9; ++j) ap[j] = e[j] * inv;
}

// ================= weighted scatter-mean per superpixel (atomic-free) =================
// NHWC=true: feat [B,H,W,C];  NHWC=false: feat [B,C,H,W] (raw inputs x / problabel)
template <int C, bool NHWC>
__global__ void spf2_k(const float* __restrict__ feat, const float* __restrict__ assoc,
                       float* __restrict__ spf)
{
  __shared__ float red[256];
  __shared__ float denTot;
  int k = blockIdx.x, b = blockIdx.y, tid = threadIdx.x;
  int ky = k / 16, kx = k % 16;
  int py = tid / 16, px = tid % 16;

  float num[C];
  #pragma unroll
  for (int c = 0; c < C; ++c) num[c] = 0.f;
  float den = 0.f;

  for (int dy2 = -1; dy2 <= 1; ++dy2)
    for (int dx2 = -1; dx2 <= 1; ++dx2) {
      int nby = ky + dy2, nbx = kx + dx2;
      if (nby < 0 || nby >= 16 || nbx < 0 || nbx >= 16) continue;   // uniform
      int y = nby * 16 + py, x = nbx * 16 + px;
      size_t p = ((size_t)b * HQ + y) * WQ + x;
      #pragma unroll
      for (int j = 0; j < 9; ++j) {
        int cy = clip16(nby + j / 3 - 1), cx = clip16(nbx + j % 3 - 1);
        if (cy == ky && cx == kx) {                                  // uniform
          float w = assoc[p * 9 + j];
          den += w;
          #pragma unroll
          for (int c = 0; c < C; ++c) {
            float fv = NHWC ? feat[p * C + c]
                            : feat[(((size_t)b * C + c) * HQ + y) * WQ + x];
            num[c] += w * fv;
          }
        }
      }
    }

  red[tid] = den; __syncthreads();
  for (int s = 128; s > 0; s >>= 1) { if (tid < s) red[tid] += red[tid + s]; __syncthreads(); }
  if (tid == 0) denTot = red[0] + EPSQ;
  __syncthreads();
  for (int c = 0; c < C; ++c) {
    red[tid] = num[c]; __syncthreads();
    for (int s = 128; s > 0; s >>= 1) { if (tid < s) red[tid] += red[tid + s]; __syncthreads(); }
    if (tid == 0) spf[(b * KSP + k) * C + c] = red[0] / denTot;
    __syncthreads();
  }
}

// ================= final: argmax labels, hard recon, soft label decode =================
__global__ void final_k(const float* __restrict__ assoc, const float* __restrict__ nspf,
                        const float* __restrict__ spl, float* __restrict__ outF,
                        float* __restrict__ outL)
{
  int idx = blockIdx.x * blockDim.x + threadIdx.x;
  if (idx >= BQ * HQ * WQ) return;
  int x = idx % WQ; int t = idx / WQ; int y = t % HQ; int b = t / HQ;
  int by = y >> 4, bx = x >> 4;

  float a[9]; int sp[9];
  const float* ap = assoc + (size_t)idx * 9;
  #pragma unroll
  for (int j = 0; j < 9; ++j) {
    sp[j] = clip16(by + j / 3 - 1) * 16 + clip16(bx + j % 3 - 1);
    a[j] = ap[j];
  }
  int best = 0; float bv = a[0];
  #pragma unroll
  for (int j = 1; j < 9; ++j) if (a[j] > bv) { bv = a[j]; best = j; }   // first-max
  int hard = sp[best];

  #pragma unroll
  for (int c = 0; c < 5; ++c)
    outF[(((size_t)b * 5 + c) * HQ + y) * WQ + x] = nspf[(b * KSP + hard) * 5 + c];

  for (int c = 0; c < 50; ++c) {
    float acc = 0.f;
    #pragma unroll
    for (int j = 0; j < 9; ++j) acc += a[j] * spl[(b * KSP + sp[j]) * 50 + c];
    outL[(((size_t)b * 50 + c) * HQ + y) * WQ + x] = acc;
  }
}

// ================================================================
extern "C" void kernel_launch(void* const* d_in, const int* in_sizes, int n_in,
                              void* d_out, int out_size, void* d_ws, size_t ws_size,
                              hipStream_t stream)
{
  (void)in_sizes; (void)n_in; (void)out_size; (void)ws_size;

  const float* x         = (const float*)d_in[0];
  const float* problabel = (const float*)d_in[5];

  // params: conv1..conv6 = (w,b,g,beta,mean,var) at indices 8.., conv7 = (w,b) at 44,45
  const float *cw[7], *cb[7], *cg[7], *cbt[7], *cm[7], *cv[7];
  int base = 8;
  for (int l = 0; l < 6; ++l) {
    cw[l]  = (const float*)d_in[base + 0]; cb[l]  = (const float*)d_in[base + 1];
    cg[l]  = (const float*)d_in[base + 2]; cbt[l] = (const float*)d_in[base + 3];
    cm[l]  = (const float*)d_in[base + 4]; cv[l]  = (const float*)d_in[base + 5];
    base += 6;
  }
  cw[6] = (const float*)d_in[44]; cb[6] = (const float*)d_in[45];
  cg[6] = cbt[6] = cm[6] = cv[6] = cb[6];   // unused when has_bn==0

  // ---- workspace partition (bf16 activation buffers contiguous for bulk zero) ----
  char* wsb = (char*)d_ws; size_t off = 0;
  auto alloc = [&](size_t bytes) -> void* {
    void* p = wsb + off; off = (off + bytes + 255) & ~(size_t)255; return p;
  };
  const size_t P = (size_t)HQ * WQ;
  const size_t P1 = 258 * 258, P2 = 130 * 130, P3 = 66 * 66;   // padded pixel counts
  size_t zeroBeg = off;
  bf16t* xp   = (bf16t*)alloc(BQ * P1 * 32 * 2);
  bf16t* c1   = (bf16t*)alloc(BQ * P1 * 64 * 2);
  bf16t* c2   = (bf16t*)alloc(BQ * P1 * 64 * 2);
  bf16t* p1   = (bf16t*)alloc(BQ * P2 * 64 * 2);
  bf16t* c3   = (bf16t*)alloc(BQ * P2 * 64 * 2);
  bf16t* c4   = (bf16t*)alloc(BQ * P2 * 64 * 2);
  bf16t* p2   = (bf16t*)alloc(BQ * P3 * 64 * 2);
  bf16t* c5   = (bf16t*)alloc(BQ * P3 * 64 * 2);
  bf16t* c6   = (bf16t*)alloc(BQ * P3 * 64 * 2);
  bf16t* cat  = (bf16t*)alloc(BQ * P1 * 224 * 2);
  size_t zeroEnd = off;
  float* feat = (float*)alloc(BQ * P * 20 * 4);          // NHWC f32 (unpadded)
  float* assoc= (float*)alloc(BQ * P * 9 * 4);           // [B,H,W,9]
  float* spf  = (float*)alloc((size_t)BQ * KSP * 20 * 4);
  float* nspf = (float*)alloc((size_t)BQ * KSP * 5 * 4);
  float* spl  = (float*)alloc((size_t)BQ * KSP * 50 * 4);

  const int Cin [7] = {   5,  64,  64,  64,  64,  64,  197};
  const int Cpad[7] = {  32,  64,  64,  64,  64,  64,  224};
  const int Cout[7] = {  64,  64,  64,  64,  64,  64,   15};
  const int CoP [7] = {  64,  64,  64,  64,  64,  64,   16};
  bf16t* wbuf[7]; float* sbuf[7]; float* tbuf[7];
  for (int l = 0; l < 7; ++l) {
    wbuf[l] = (bf16t*)alloc((size_t)CoP[l] * 9 * Cpad[l] * 2);
    sbuf[l] = (float*)alloc((size_t)CoP[l] * 4);
    tbuf[l] = (float*)alloc((size_t)CoP[l] * 4);
  }

  // ---- zero all padded activation buffers (halos must be zero) ----
  {
    int n16 = (int)((zeroEnd - zeroBeg) / 16);
    zero_k<<<(n16 + 255) / 256, 256, 0, stream>>>((uint4*)(wsb + zeroBeg), n16);
  }

  // ---- weight prep ----
  for (int l = 0; l < 7; ++l) {
    int tot = CoP[l] * 9 * Cpad[l];
    prep_weights_k<<<(tot + 255) / 256, 256, 0, stream>>>(cw[l], wbuf[l], Cout[l], Cin[l],
                                                          Cpad[l], CoP[l]);
    prep_scalebias_k<<<1, 64, 0, stream>>>(cb[l], cg[l], cbt[l], cm[l], cv[l],
                                           sbuf[l], tbuf[l], Cout[l], CoP[l], l < 6 ? 1 : 0);
  }

  // ---- pack input ----
  const int gpix = (int)((BQ * P + 255) / 256);
  pack_x_k<<<gpix, 256, 0, stream>>>(x, xp);

  // ---- CNN backbone (implicit-GEMM WMMA convs, padded bf16 NHWC) ----
  #define CONV(L, CPAD, HH, WW, IN, OUT, RELU, OFF, TOT, F32)                         \
    conv3x3_wmma_k<CPAD, HH, WW, F32>                                                 \
        <<<dim3(((HH) * (WW)) / 16, CoP[L] / 16, BQ), 32, 0, stream>>>(               \
            IN, wbuf[L], sbuf[L], tbuf[L], (void*)(OUT), Cout[L], RELU, OFF, TOT)

  CONV(0,  32, 256, 256, xp, c1, 1, 0, 64, false);
  CONV(1,  64, 256, 256, c1, c2, 1, 0, 64, false);
  maxpool_k<256, 256><<<(int)((BQ * 128 * 128 * 64 + 255) / 256), 256, 0, stream>>>(c2, p1);
  CONV(2,  64, 128, 128, p1, c3, 1, 0, 64, false);
  CONV(3,  64, 128, 128, c3, c4, 1, 0, 64, false);
  maxpool_k<128, 128><<<(int)((BQ * 64 * 64 * 64 + 255) / 256), 256, 0, stream>>>(c4, p2);
  CONV(4,  64,  64,  64, p2, c5, 1, 0, 64, false);
  CONV(5,  64,  64,  64, c5, c6, 1, 0, 64, false);
  build_cat_k<<<gpix, 256, 0, stream>>>(x, c2, c4, c6, cat);
  CONV(6, 224, 256, 256, cat, feat, 0, 5, 20, true);   // conv7 -> feat[:,:,:,5:20] f32
  copyx_k<<<(int)((BQ * P * 5 + 255) / 256), 256, 0, stream>>>(x, feat);
  #undef CONV

  // ---- SSN iterations ----
  dim3 gsp(KSP, BQ);
  spf_init_k<<<gsp, 256, 0, stream>>>(feat, spf);
  for (int it = 0; it < NITER; ++it) {
    assign_k<<<gpix, 256, 0, stream>>>(feat, spf, assoc);
    spf2_k<20, true><<<gsp, 256, 0, stream>>>(feat, assoc, spf);
  }
  assign_k<<<gpix, 256, 0, stream>>>(feat, spf, assoc);
  spf2_k<5,  false><<<gsp, 256, 0, stream>>>(x,         assoc, nspf);
  spf2_k<50, false><<<gsp, 256, 0, stream>>>(problabel, assoc, spl);

  // ---- outputs: [B,5,H,W] then [B,50,H,W] f32 ----
  float* outF = (float*)d_out;
  float* outL = outF + BQ * 5 * P;
  final_k<<<gpix, 256, 0, stream>>>(assoc, nspf, spl, outF, outL);
}
